// SelfAttentionLayer_11536282157336
// MI455X (gfx1250) — compile-verified, matrix operand order checked by reading, not run on hardware
//
#include <hip/hip_runtime.h>
#include <hip/hip_bf16.h>

// Problem constants (reference: B,N,C,D = 16,2048,256,256)
constexpr int B = 16;
constexpr int N = 2048;
constexpr int C = 256;
constexpr int D = 256;

typedef __attribute__((ext_vector_type(8)))  float  v8f;
typedef __attribute__((ext_vector_type(16))) __bf16 v16bf;
typedef __attribute__((ext_vector_type(8)))  __bf16 v8bf;
typedef int v4i __attribute__((vector_size(16)));

#if defined(__has_builtin)
#  if __has_builtin(__builtin_amdgcn_global_load_async_to_lds_b128)
#    define ASYNC_LDS 1
#  endif
#endif

// fp32 -> bf16 with round-to-nearest-even (pure bit math)
static __device__ inline __bf16 f2bf(float f) {
    unsigned u = __builtin_bit_cast(unsigned, f);
    u = u + 0x7FFFu + ((u >> 16) & 1u);
    unsigned short s = (unsigned short)(u >> 16);
    return __builtin_bit_cast(__bf16, s);
}

static __device__ inline v8f wmma_bf16(v16bf a, v16bf b, v8f c) {
    return __builtin_amdgcn_wmma_f32_16x16x32_bf16(
        false, a, false, b, (short)0, c, false, false);
}

// A-fragment (16x32, 16-bit) loader from a row-major bf16 matrix (global or LDS).
// Per ISA: lanes 0-15: row=lane, K={0..7}U{16..23}; lanes 16-31: row=lane-16, K={8..15}U{24..31}.
static __device__ inline v16bf frag_bf16(const __bf16* __restrict__ base, int ld,
                                         int row_base, int k_base, int lane) {
    const int r  = row_base + (lane & 15);
    const int hi = (lane >> 4) & 1;
    const __bf16* p = base + (size_t)r * ld + k_base + hi * 8;
    v8bf lo = *(const v8bf*)p;
    v8bf hh = *(const v8bf*)(p + 16);
    return __builtin_shufflevector(lo, hh, 0, 1, 2, 3, 4, 5, 6, 7,
                                   8, 9, 10, 11, 12, 13, 14, 15);
}

// 16-byte global -> LDS copy: async-to-LDS on CDNA5, plain copy fallback.
static __device__ inline void cp16(const __bf16* g, __bf16* l) {
#ifdef ASYNC_LDS
    __builtin_amdgcn_global_load_async_to_lds_b128(
        (__attribute__((address_space(1))) v4i*)(g),
        (__attribute__((address_space(3))) v4i*)(l), 0, 0);
#else
    *(v8bf*)l = *(const v8bf*)g;
#endif
}
// Wait until at most NN async ops remain outstanding (in-order completion per wave).
template <int NN> static __device__ inline void async_wait() {
#ifdef ASYNC_LDS
#  if __has_builtin(__builtin_amdgcn_s_wait_asynccnt)
    __builtin_amdgcn_s_wait_asynccnt(NN);
#  else
    asm volatile("s_wait_asynccnt %0" :: "i"(NN) : "memory");
#  endif
#endif
}

// ---------------------------------------------------------------------------
// Kernel 0: streaming fp32 -> bf16 conversion (8 elements / thread)
// ---------------------------------------------------------------------------
__global__ __launch_bounds__(256) void to_bf16(const float* __restrict__ src,
                                               __bf16* __restrict__ dst, int n8) {
    const int i = blockIdx.x * 256 + threadIdx.x;
    if (i >= n8) return;
    const float4* p = (const float4*)src + (size_t)i * 2;
    float4 a = p[0], b = p[1];
    v8bf o;
    o[0] = f2bf(a.x); o[1] = f2bf(a.y); o[2] = f2bf(a.z); o[3] = f2bf(a.w);
    o[4] = f2bf(b.x); o[5] = f2bf(b.y); o[6] = f2bf(b.z); o[7] = f2bf(b.w);
    *((v8bf*)dst + i) = o;
}

// ---------------------------------------------------------------------------
// Kernel 1: QKV projection from bf16 inputs.
// Q,K -> bf16 [B*N, D] row-major; V -> bf16 Vt[B, D, N].
// grid = (B*N/16, D/64), block = 128 (4 waves); wave owns one 16x16 tile.
// ---------------------------------------------------------------------------
__global__ __launch_bounds__(128) void qkv_proj(
    const __bf16* __restrict__ xb,
    const __bf16* __restrict__ Wqb, const float* __restrict__ bq,
    const __bf16* __restrict__ Wkb, const float* __restrict__ bk,
    const __bf16* __restrict__ Wvb, const float* __restrict__ bv,
    __bf16* __restrict__ Q, __bf16* __restrict__ Kmat, __bf16* __restrict__ Vt)
{
    const int lane = threadIdx.x & 31;
    const int wave = threadIdx.x >> 5;
    const int g0 = blockIdx.x * 16;          // global token row base
    const int b  = g0 / N;
    const int n0 = g0 - b * N;
    const int d0 = (blockIdx.y * 4 + wave) * 16;

    v8f aq = {}, ak = {}, av = {};
#pragma unroll
    for (int kc = 0; kc < C; kc += 32) {
        v16bf ax = frag_bf16(xb,  C, g0, kc, lane);  // A: x tile [16 tok x 32 c]
        v16bf wq = frag_bf16(Wqb, C, d0, kc, lane);  // B^T: W rows [16 d x 32 c]
        v16bf wk = frag_bf16(Wkb, C, d0, kc, lane);
        v16bf wv = frag_bf16(Wvb, C, d0, kc, lane);
        aq = wmma_bf16(ax, wq, aq);
        ak = wmma_bf16(ax, wk, ak);
        av = wmma_bf16(ax, wv, av);
    }

    const int col = lane & 15;
    const int hi  = (lane >> 4) & 1;
    const int dcol = d0 + col;
    const float biq = bq[dcol], bik = bk[dcol], biv = bv[dcol];

#pragma unroll
    for (int i = 0; i < 8; ++i) {
        const size_t g = (size_t)(g0 + i + 8 * hi);
        Q[g * D + dcol]    = f2bf(aq[i] + biq);
        Kmat[g * D + dcol] = f2bf(ak[i] + bik);
    }
    v8bf pv;
#pragma unroll
    for (int i = 0; i < 8; ++i) pv[i] = f2bf(av[i] + biv);
    *(v8bf*)(Vt + ((size_t)b * D + dcol) * N + n0 + 8 * hi) = pv;
}

// ---------------------------------------------------------------------------
// Kernel 2: masked flash attention, double-buffered async K/V staging in LDS.
// grid = (N/64, B), block = 128 (4 waves); each wave owns a 16-query tile.
// ---------------------------------------------------------------------------
constexpr int KLD   = 264;            // LDS row stride (halves), K rows of 256
constexpr int VLD   = 40;             // LDS row stride (halves), V rows of 32
constexpr int KTILE = 32 * KLD;       // halves per K tile
constexpr int VTILE = 256 * VLD;      // halves per V tile
constexpr int BUFSZ = KTILE + VTILE;  // halves per (K,V) buffer
constexpr int PSTG  = 16 * 40;        // halves per wave P-stage
constexpr size_t SMEM_BYTES = (size_t)(2 * BUFSZ + 4 * PSTG) * sizeof(__bf16); // ~78 KB

__global__ __launch_bounds__(128) void flash_attn(
    const __bf16* __restrict__ Q, const __bf16* __restrict__ Kmat,
    const __bf16* __restrict__ Vt, const int* __restrict__ valid_len,
    float* __restrict__ out)
{
    extern __shared__ __align__(16) __bf16 smem[];
    __bf16* buf0 = smem;
    __bf16* buf1 = smem + BUFSZ;

    const int tid  = threadIdx.x;
    const int lane = tid & 31;
    const int wave = tid >> 5;
    const int b  = blockIdx.y;
    const int q0 = blockIdx.x * 64 + wave * 16;
    const int col = lane & 15;
    const int hi  = (lane >> 4) & 1;
    const int vlen = valid_len[b];

    const __bf16* Qb  = Q    + (size_t)b * N * D;
    const __bf16* Kb  = Kmat + (size_t)b * N * D;
    const __bf16* Vtb = Vt   + (size_t)b * D * N;
    __bf16* st = smem + 2 * BUFSZ + wave * PSTG;

    // stage one 32-key block: K [32 x 256] + V [256 x 32] -> 16 x 16B chunks/thread
    auto stage = [&](int kb, __bf16* dst) {
        __bf16* kd = dst;
        __bf16* vd = dst + KTILE;
#pragma unroll
        for (int i = 0; i < 8; ++i) {
            const int c = tid + i * 128;
            cp16(Kb + (size_t)(kb + (c >> 5)) * D + (c & 31) * 8,
                 kd + (c >> 5) * KLD + (c & 31) * 8);
        }
#pragma unroll
        for (int i = 0; i < 8; ++i) {
            const int c = tid + i * 128;
            cp16(Vtb + (size_t)(c >> 2) * N + kb + (c & 3) * 8,
                 vd + (c >> 2) * VLD + (c & 3) * 8);
        }
    };

    // Preload this wave's 8 Q A-fragments (16 x 256)
    v16bf qf[8];
#pragma unroll
    for (int kc = 0; kc < 8; ++kc) qf[kc] = frag_bf16(Qb, D, q0, kc * 32, lane);

    float m[8], l[8];
#pragma unroll
    for (int i = 0; i < 8; ++i) { m[i] = -3.0e38f; l[i] = 0.0f; }
    v8f o[16];
#pragma unroll
    for (int t = 0; t < 16; ++t) o[t] = {};

    stage(0, buf0);  // prologue: first tile in flight

    for (int kb = 0; kb < N; kb += 32) {
        const int curi = (kb >> 5) & 1;
        __bf16* kbuf = curi ? buf1 : buf0;
        __bf16* vbuf = kbuf + KTILE;

        if (kb + 32 < N) {
            // issue next tile into the other buffer (its readers finished last iter,
            // guarded by the trailing __syncthreads), then wait for *current* tile:
            // async ops complete in order, so <=16 outstanding => current 16 done.
            stage(kb + 32, curi ? buf0 : buf1);
            async_wait<16>();
        } else {
            async_wait<0>();
        }
        __syncthreads();

        // ---- scores: two 16x16 tiles from LDS-resident K ----
        v8f s0 = {}, s1 = {};
#pragma unroll
        for (int kc = 0; kc < 8; ++kc) {
            v16bf k0 = frag_bf16(kbuf, KLD, 0,  kc * 32, lane);
            v16bf k1 = frag_bf16(kbuf, KLD, 16, kc * 32, lane);
            s0 = wmma_bf16(qf[kc], k0, s0);
            s1 = wmma_bf16(qf[kc], k1, s1);
        }

        const bool ok0 = (kb + col)      < vlen;
        const bool ok1 = (kb + 16 + col) < vlen;

        float scale[8];
#pragma unroll
        for (int i = 0; i < 8; ++i) {
            float a0 = ok0 ? s0[i] : -1.0e6f;
            float a1 = ok1 ? s1[i] : -1.0e6f;
            float rm = fmaxf(a0, a1);
#pragma unroll
            for (int off = 1; off < 16; off <<= 1) rm = fmaxf(rm, __shfl_xor(rm, off, 32));
            float mn = fmaxf(m[i], rm);
            float sc = __expf(m[i] - mn);
            float p0 = __expf(a0 - mn);
            float p1 = __expf(a1 - mn);
            float rs = p0 + p1;
#pragma unroll
            for (int off = 1; off < 16; off <<= 1) rs += __shfl_xor(rs, off, 32);
            l[i] = l[i] * sc + rs;
            m[i] = mn;
            scale[i] = sc;
            const int r = i + 8 * hi;
            st[r * 40 + col]      = f2bf(p0);
            st[r * 40 + 16 + col] = f2bf(p1);
        }

#pragma unroll
        for (int t = 0; t < 16; ++t) {
#pragma unroll
            for (int i = 0; i < 8; ++i) o[t][i] *= scale[i];
        }

        __builtin_amdgcn_sched_barrier(0);   // ds_store before ds_load; DS in-order per wave
        const __bf16* pp = st + (lane & 15) * 40 + hi * 8;
        v8bf plo = *(const v8bf*)pp;
        v8bf phh = *(const v8bf*)(pp + 16);
        v16bf pf = __builtin_shufflevector(plo, phh, 0, 1, 2, 3, 4, 5, 6, 7,
                                           8, 9, 10, 11, 12, 13, 14, 15);
        __builtin_amdgcn_sched_barrier(0);

        // ---- O += P(16x32) * V(32x16) per 16-wide d tile, V from LDS ----
#pragma unroll
        for (int t = 0; t < 16; ++t) {
            v16bf vf = frag_bf16(vbuf, VLD, t * 16, 0, lane);
            o[t] = wmma_bf16(pf, vf, o[t]);
        }

        __syncthreads();   // all waves done with this buffer before it is restaged
    }

    float inv[8];
#pragma unroll
    for (int i = 0; i < 8; ++i) inv[i] = 1.0f / l[i];
#pragma unroll
    for (int t = 0; t < 16; ++t) {
        const int d = t * 16 + col;
#pragma unroll
        for (int i = 0; i < 8; ++i) {
            const int n = q0 + i + 8 * hi;
            out[((size_t)b * N + n) * D + d] = o[t][i] * inv[i];
        }
    }
}

extern "C" void kernel_launch(void* const* d_in, const int* in_sizes, int n_in,
                              void* d_out, int out_size, void* d_ws, size_t ws_size,
                              hipStream_t stream) {
    const float* x    = (const float*)d_in[0];
    const int*   vlen = (const int*)  d_in[1];
    const float* Wq   = (const float*)d_in[2];
    const float* bq   = (const float*)d_in[3];
    const float* Wk   = (const float*)d_in[4];
    const float* bk   = (const float*)d_in[5];
    const float* Wv   = (const float*)d_in[6];
    const float* bv   = (const float*)d_in[7];
    float* out = (float*)d_out;

    // Workspace layout (bf16): Q, K [B*N,D]; Vt [B,D,N]; xb [B*N,C]; Wq/Wk/Wv [D,C]
    __bf16* Qw  = (__bf16*)d_ws;
    __bf16* Kw  = Qw + (size_t)B * N * D;
    __bf16* Vtw = Kw + (size_t)B * N * D;
    __bf16* xb  = Vtw + (size_t)B * N * D;
    __bf16* Wqb = xb + (size_t)B * N * C;
    __bf16* Wkb = Wqb + (size_t)D * C;
    __bf16* Wvb = Wkb + (size_t)D * C;

    // Allow >64KB dynamic LDS (ISA permits up to 320KB/workgroup on CDNA5).
    (void)hipFuncSetAttribute((const void*)flash_attn,
                              hipFuncAttributeMaxDynamicSharedMemorySize,
                              (int)SMEM_BYTES);

    const int nx8 = B * N * C / 8;
    const int nw8 = D * C / 8;
    to_bf16<<<dim3((nx8 + 255) / 256), 256, 0, stream>>>(x,  xb,  nx8);
    to_bf16<<<dim3((nw8 + 255) / 256), 256, 0, stream>>>(Wq, Wqb, nw8);
    to_bf16<<<dim3((nw8 + 255) / 256), 256, 0, stream>>>(Wk, Wkb, nw8);
    to_bf16<<<dim3((nw8 + 255) / 256), 256, 0, stream>>>(Wv, Wvb, nw8);

    qkv_proj<<<dim3(B * N / 16, D / 64), 128, 0, stream>>>(
        xb, Wqb, bq, Wkb, bk, Wvb, bv, Qw, Kw, Vtw);
    flash_attn<<<dim3(N / 64, B), 128, SMEM_BYTES, stream>>>(Qw, Kw, Vtw, vlen, out);
}
